// FEGIN_68899865362614
// MI455X (gfx1250) — matrix operand order repaired and checked by reading, not compiled, not run on hardware
//
#include <hip/hip_runtime.h>
#include <hip/hip_bf16.h>

// ---------------------------------------------------------------------------
// Types for CDNA5 WMMA (wave32)
// ---------------------------------------------------------------------------
typedef __attribute__((ext_vector_type(16))) __bf16 v16bf;
typedef __attribute__((ext_vector_type(8)))  __bf16 v8bf;
typedef __attribute__((ext_vector_type(8)))  float  v8f;

union Frag16 { v16bf v; v8bf h[2]; };

// ---------------------------------------------------------------------------
// Small utility kernels
// ---------------------------------------------------------------------------
__global__ void zero_f32(float* p, int n) {
    int i = blockIdx.x * blockDim.x + threadIdx.x;
    if (i < n) p[i] = 0.0f;
}

__global__ void count_nodes(const long long* __restrict__ batch, float* cnt, int n) {
    int i = blockIdx.x * blockDim.x + threadIdx.x;
    if (i < n) atomicAdd(&cnt[(int)batch[i]], 1.0f);
}

// agg = (1 + eps[l]) * x   (vectorized float4)
__global__ void init_agg(const float* __restrict__ x, const float* __restrict__ eps,
                         int l, float* __restrict__ agg, int n4) {
    int i = blockIdx.x * blockDim.x + threadIdx.x;
    if (i >= n4) return;
    float s = 1.0f + eps[l];
    float4 v = ((const float4*)x)[i];
    float4 o; o.x = v.x * s; o.y = v.y * s; o.z = v.z * s; o.w = v.w * s;
    ((float4*)agg)[i] = o;
}

// agg[dst] += x[src] : one thread per (edge, 4-feature chunk)
__global__ void edge_agg(const float* __restrict__ x, const long long* __restrict__ ei,
                         float* __restrict__ agg, int E) {
    int t = blockIdx.x * blockDim.x + threadIdx.x;
    if (t >= E * 32) return;
    int e = t >> 5;
    int c = (t & 31) * 4;
    int s = (int)ei[e];         // src
    int d = (int)ei[E + e];     // dst
    float4 v = *(const float4*)(x + (size_t)s * 128 + c);
    float* dp = agg + (size_t)d * 128 + c;
    atomicAdd(dp + 0, v.x);
    atomicAdd(dp + 1, v.y);
    atomicAdd(dp + 2, v.z);
    atomicAdd(dp + 3, v.w);
}

__global__ void cvt_bf16(const float* __restrict__ in, __bf16* __restrict__ out, int n) {
    int i = blockIdx.x * blockDim.x + threadIdx.x;
    if (i < n) out[i] = (__bf16)in[i];
}

// Pack one 128x128 f32 weight matrix (row = K, col = N) into per-lane bf16
// B-fragments for v_wmma_f32_16x16x32_bf16:
//   packed[((kt*8 + ntile)*32 + lane)*16 + i] = W[kt*32 + (lane>>4)*8 + (i&7) + (i>>3)*16][ntile*16 + (lane&15)]
__global__ void pack_w(const float* __restrict__ W, __bf16* __restrict__ out) {
    int t = blockIdx.x * blockDim.x + threadIdx.x;
    if (t >= 128 * 128) return;
    int i     =  t        & 15;
    int lane  = (t >> 4)  & 31;
    int ntile = (t >> 9)  & 7;
    int kt    = (t >> 12) & 3;
    int col = ntile * 16 + (lane & 15);
    int k   = kt * 32 + (lane >> 4) * 8 + (i & 7) + (i >> 3) * 16;
    out[t] = (__bf16)W[k * 128 + col];
}

// ---------------------------------------------------------------------------
// WMMA GEMM: O[nrows,128] = relu(A[nrows,128] @ W[128,128] + bias)
// Block = 256 threads = 8 waves, covering 64 rows x 128 cols.
// Wave w: row tile (w&3), column half (w>>2) -> 4 WMMA fragments per wave.
// ---------------------------------------------------------------------------
template <bool OUT_BF16>
__global__ __launch_bounds__(256)
void gemm128_bf16(const __bf16* __restrict__ A, const __bf16* __restrict__ Bp,
                  const float* __restrict__ bias, __bf16* __restrict__ obf,
                  float* __restrict__ of32, int nrows) {
    const int lane = threadIdx.x & 31;
    const int w    = threadIdx.x >> 5;
    const int rt   = w & 3;
    const int ch   = w >> 2;
    const int rowbase = blockIdx.x * 64 + rt * 16;

    int arow = rowbase + (lane & 15);
    if (arow >= nrows) arow = nrows - 1;          // clamp (stores are guarded)
    const __bf16* Ar = A + (size_t)arow * 128;
    const int kb = (lane >> 4) * 8;               // K sub-offset per half-wave

    v8f acc[4] = {};
#pragma unroll
    for (int kt = 0; kt < 4; ++kt) {
        Frag16 a;
        a.h[0] = *(const v8bf*)(Ar + kt * 32 + kb);        // K = kb..kb+7
        a.h[1] = *(const v8bf*)(Ar + kt * 32 + kb + 16);   // K = kb+16..kb+23
#pragma unroll
        for (int n = 0; n < 4; ++n) {
            const int ntile = ch * 4 + n;
            const __bf16* bp = Bp + ((size_t)((kt * 8 + ntile) * 32 + lane)) * 16;
            Frag16 b;
            b.h[0] = *(const v8bf*)(bp);
            b.h[1] = *(const v8bf*)(bp + 8);
            acc[n] = __builtin_amdgcn_wmma_f32_16x16x32_bf16(
                false, a.v, false, b.v, (short)0, acc[n], false, false);
        }
    }

    const int r0 = rowbase + (lane >> 4) * 8;     // C layout: VGPR r -> row r (+8 hi half)
#pragma unroll
    for (int n = 0; n < 4; ++n) {
        const int col = ch * 64 + n * 16 + (lane & 15);
        const float bv = bias[col];
#pragma unroll
        for (int r = 0; r < 8; ++r) {
            const int row = r0 + r;
            if (row < nrows) {
                float v = acc[n][r] + bv;
                v = fmaxf(v, 0.0f);
                if (OUT_BF16) obf[(size_t)row * 128 + col] = (__bf16)v;
                else          of32[(size_t)row * 128 + col] = v;
            }
        }
    }
}

// ---------------------------------------------------------------------------
// BatchNorm: per-channel sum / sumsq via LDS ds_add_f32 then global atomics
// ---------------------------------------------------------------------------
__global__ void bn_stats(const float* __restrict__ h, float* sum, float* sq, int n) {
    __shared__ float ls[128];
    __shared__ float lq[128];
    if (threadIdx.x < 128) { ls[threadIdx.x] = 0.0f; lq[threadIdx.x] = 0.0f; }
    __syncthreads();
    for (int i = blockIdx.x * blockDim.x + threadIdx.x; i < n;
         i += gridDim.x * blockDim.x) {
        float v = h[i];
        int c = i & 127;
        atomicAdd(&ls[c], v);
        atomicAdd(&lq[c], v * v);
    }
    __syncthreads();
    if (threadIdx.x < 128) {
        atomicAdd(&sum[threadIdx.x], ls[threadIdx.x]);
        atomicAdd(&sq[threadIdx.x],  lq[threadIdx.x]);
    }
}

__global__ void bn_finalize(const float* sum, const float* sq,
                            const float* __restrict__ g, const float* __restrict__ b,
                            float* scale, float* shift, int n) {
    int c = threadIdx.x;
    if (c >= 128) return;
    float inv  = 1.0f / (float)n;
    float mean = sum[c] * inv;
    float var  = fmaxf(sq[c] * inv - mean * mean, 0.0f);
    float sc   = g[c] * rsqrtf(var + 1e-5f);
    scale[c] = sc;
    shift[c] = b[c] - mean * sc;
}

// x_next = h*scale + shift ; also pooled[batch[row], l*128+c] += x_next
__global__ void bn_apply(const float* __restrict__ h, const float* __restrict__ scale,
                         const float* __restrict__ shift,
                         const long long* __restrict__ batch,
                         float* __restrict__ xnext, float* __restrict__ pooled,
                         int l, int n) {
    int i = blockIdx.x * blockDim.x + threadIdx.x;
    if (i >= n) return;
    int row = i >> 7;
    int c   = i & 127;
    float v = h[i] * scale[c] + shift[c];
    xnext[i] = v;
    atomicAdd(&pooled[(int)batch[row] * 512 + l * 128 + c], v);
}

// ---------------------------------------------------------------------------
// Head (tiny, f32)
// ---------------------------------------------------------------------------
__global__ void zbuild(const float* __restrict__ pooled, const float* __restrict__ cnt,
                       const float* __restrict__ emb, float* __restrict__ z, int G) {
    int t = blockIdx.x * blockDim.x + threadIdx.x;
    if (t >= G * 576) return;
    int g = t / 576, c = t % 576;
    float v;
    if (c < 512) v = pooled[g * 512 + c] / fmaxf(cnt[g], 1.0f);
    else         v = emb[g * 64 + (c - 512)];
    z[t] = v;
}

__global__ void dense_relu(const float* __restrict__ in, const float* __restrict__ W,
                           const float* __restrict__ b, float* __restrict__ out,
                           int M, int K, int Nc) {
    int t = blockIdx.x * blockDim.x + threadIdx.x;
    if (t >= M * Nc) return;
    int m = t / Nc, n = t % Nc;
    float s = b[n];
    for (int k = 0; k < K; ++k) s += in[m * K + k] * W[k * Nc + n];
    out[t] = fmaxf(s, 0.0f);
}

__global__ void head_logsoftmax(const float* __restrict__ t2, const float* __restrict__ w4,
                                const float* __restrict__ b4, float* __restrict__ out,
                                int G) {
    int g = blockIdx.x * blockDim.x + threadIdx.x;
    if (g >= G) return;
    float logit[10];
    for (int c = 0; c < 10; ++c) {
        float s = b4[c];
        for (int k = 0; k < 128; ++k) s += t2[g * 128 + k] * w4[k * 10 + c];
        logit[c] = s;
    }
    float m = logit[0];
    for (int c = 1; c < 10; ++c) m = fmaxf(m, logit[c]);
    float se = 0.0f;
    for (int c = 0; c < 10; ++c) se += __expf(logit[c] - m);
    float lse = __logf(se);
    for (int c = 0; c < 10; ++c) out[g * 10 + c] = logit[c] - m - lse;
}

// ---------------------------------------------------------------------------
// Launch
// ---------------------------------------------------------------------------
extern "C" void kernel_launch(void* const* d_in, const int* in_sizes, int n_in,
                              void* d_out, int out_size, void* d_ws, size_t ws_size,
                              hipStream_t stream) {
    const float*     x0    = (const float*)d_in[0];
    const long long* ei    = (const long long*)d_in[1];
    const long long* batch = (const long long*)d_in[2];
    const float*     emb   = (const float*)d_in[3];
    const float*     eps   = (const float*)d_in[4];
    const float*     W1    = (const float*)d_in[5];
    const float*     b1    = (const float*)d_in[6];
    const float*     W2    = (const float*)d_in[7];
    const float*     b2    = (const float*)d_in[8];
    const float*     gamma = (const float*)d_in[9];
    const float*     beta  = (const float*)d_in[10];
    const float*     l1w   = (const float*)d_in[11];
    const float*     l1b   = (const float*)d_in[12];
    const float*     l2w   = (const float*)d_in[13];
    const float*     l2b   = (const float*)d_in[14];
    const float*     l4w   = (const float*)d_in[15];
    const float*     l4b   = (const float*)d_in[16];
    float*           out   = (float*)d_out;

    const int F = 128, H = 128, L = 4, G = 64;
    const int N = in_sizes[0] / F;
    const int E = in_sizes[1] / 2;

    // Workspace carve-out (256B aligned)
    char* p = (char*)d_ws;
    auto alloc = [&](size_t b) -> char* {
        char* r = p;
        p += (b + 255) & ~(size_t)255;
        return r;
    };
    float*  xcur   = (float*) alloc((size_t)N * F * 4);
    float*  agg    = (float*) alloc((size_t)N * F * 4);
    __bf16* a0     = (__bf16*)alloc((size_t)N * F * 2);
    __bf16* a1     = (__bf16*)alloc((size_t)N * H * 2);
    float*  hbuf   = (float*) alloc((size_t)N * H * 4);
    __bf16* wpk    = (__bf16*)alloc((size_t)L * 2 * 128 * 128 * 2);
    float*  bnsum  = (float*) alloc(128 * 4);
    float*  bnsq   = (float*) alloc(128 * 4);
    float*  scale  = (float*) alloc(128 * 4);
    float*  shift  = (float*) alloc(128 * 4);
    float*  pooled = (float*) alloc((size_t)G * 512 * 4);
    float*  cnt    = (float*) alloc((size_t)G * 4);
    float*  zb     = (float*) alloc((size_t)G * 576 * 4);
    float*  t1     = (float*) alloc((size_t)G * 256 * 4);
    float*  t2     = (float*) alloc((size_t)G * 128 * 4);

    const int TB = 256;

    // Global pooling accumulators + per-graph node counts
    zero_f32<<<(G * 512 + TB - 1) / TB, TB, 0, stream>>>(pooled, G * 512);
    zero_f32<<<1, TB, 0, stream>>>(cnt, G);
    count_nodes<<<(N + TB - 1) / TB, TB, 0, stream>>>(batch, cnt, N);

    // Pack weights to WMMA B-fragment layout (bf16)
    for (int l = 0; l < L; ++l) {
        pack_w<<<64, TB, 0, stream>>>(W1 + (size_t)l * 128 * 128, wpk + (size_t)(2 * l) * 16384);
        pack_w<<<64, TB, 0, stream>>>(W2 + (size_t)l * 128 * 128, wpk + (size_t)(2 * l + 1) * 16384);
    }

    const float* xin = x0;
    for (int l = 0; l < L; ++l) {
        // agg = (1+eps)*x ; agg[dst] += x[src]
        init_agg<<<(N * 32 + TB - 1) / TB, TB, 0, stream>>>(xin, eps, l, agg, N * 32);
        edge_agg<<<(E * 32 + TB - 1) / TB, TB, 0, stream>>>(xin, ei, agg, E);
        cvt_bf16<<<(N * 128 + TB - 1) / TB, TB, 0, stream>>>(agg, a0, N * 128);

        // MLP: relu(A@W1+b1) -> relu(@W2+b2)
        gemm128_bf16<true><<<(N + 63) / 64, TB, 0, stream>>>(
            a0, wpk + (size_t)(2 * l) * 16384, b1 + l * 128, a1, nullptr, N);
        gemm128_bf16<false><<<(N + 63) / 64, TB, 0, stream>>>(
            a1, wpk + (size_t)(2 * l + 1) * 16384, b2 + l * 128, nullptr, hbuf, N);

        // BatchNorm (batch statistics)
        zero_f32<<<1, TB, 0, stream>>>(bnsum, 128);
        zero_f32<<<1, TB, 0, stream>>>(bnsq, 128);
        bn_stats<<<512, TB, 0, stream>>>(hbuf, bnsum, bnsq, N * 128);
        bn_finalize<<<1, 128, 0, stream>>>(bnsum, bnsq, gamma + l * 128, beta + l * 128,
                                           scale, shift, N);
        bn_apply<<<(N * 128 + TB - 1) / TB, TB, 0, stream>>>(
            hbuf, scale, shift, batch, xcur, pooled, l, N * 128);

        xin = xcur;
    }

    // Head: pool -> concat emb -> lin1+relu -> lin2+relu -> lin4 -> log_softmax
    zbuild<<<(G * 576 + TB - 1) / TB, TB, 0, stream>>>(pooled, cnt, emb, zb, G);
    dense_relu<<<(G * 256 + TB - 1) / TB, TB, 0, stream>>>(zb, l1w, l1b, t1, G, 576, 256);
    dense_relu<<<(G * 128 + TB - 1) / TB, TB, 0, stream>>>(t1, l2w, l2b, t2, G, 256, 128);
    head_logsoftmax<<<1, 64, 0, stream>>>(t2, l4w, l4b, out, G);
}
